// model1_42992622633178
// MI455X (gfx1250) — compile-verified, matrix operand order checked by reading, not run on hardware
//
#include <hip/hip_runtime.h>

#define HID    51
#define HP     64            // padded hidden
#define TSTEPS 1024
#define BATCH  1024
#define MTILE  16            // batch rows per block (WMMA M)
#define HSTR   72            // padded LDS f16 row stride (bank-conflict dodge)

typedef __attribute__((ext_vector_type(16))) _Float16 v16h;
typedef __attribute__((ext_vector_type(8)))  float    v8f;

// K index inside a 16-bit 16x32 A/B fragment for element e (0..15) of the v16h,
// per CDNA5 ISA §7.12.2: VGPR v holds K = {2v,2v+1} (+8 for lanes 16..31,
// +16 for VGPRs 4..7).
__device__ __forceinline__ int frag_k(int half, int e) {
  int v = e >> 1;
  return (half << 3) + ((v & 3) << 1) + ((v >> 2) << 4) + (e & 1);
}

// Fast activations: v_exp_f32 (2^x) + v_rcp_f32, no IEEE divide chains.
// Limits are exact: sigm(+-inf)=1/0, tanh(+-inf)=+-1.
#define LOG2E 1.442695040888963f
__device__ __forceinline__ float sigm(float x) {
  return __builtin_amdgcn_rcpf(1.0f + __builtin_amdgcn_exp2f(-LOG2E * x));
}
__device__ __forceinline__ float tanh_fast(float x) {
  return 1.0f - 2.0f * __builtin_amdgcn_rcpf(1.0f + __builtin_amdgcn_exp2f(2.0f * LOG2E * x));
}

union FragBits { v16h h; int i[8]; };

__device__ __forceinline__ v16h load_frag_lds(const int* base_dw, int fid, int lane) {
  FragBits u;
  const int base = (fid * 32 + lane) * 8;
  #pragma unroll
  for (int d = 0; d < 8; ++d) u.i[d] = base_dw[base + d];   // 2x ds_load_b128
  return u.h;
}

__global__ void __launch_bounds__(128)
lstm2_wmma_kernel(const float* __restrict__ x,
                  const float* __restrict__ Wih0, const float* __restrict__ Whh0,
                  const float* __restrict__ bih0, const float* __restrict__ bhh0,
                  const float* __restrict__ Wih1, const float* __restrict__ Whh1,
                  const float* __restrict__ bih1, const float* __restrict__ bhh1,
                  const float* __restrict__ Wlin, const float* __restrict__ blin,
                  float* __restrict__ out)
{
  // dynamic LDS: layer-1 weight fragments, pre-swizzled per (wave,s,kc,lane)
  extern __shared__ int B1lds[];           // 64 frags * 32 lanes * 8 dw = 64KB
  __shared__ _Float16 hA0[MTILE * HSTR];   // h0 state, fp16
  __shared__ _Float16 hA1[MTILE * HSTR];   // h1 state
  __shared__ float    xbuf[MTILE];
  __shared__ float    wlinS[HP];

  const int tid     = threadIdx.x;
  const int w       = tid >> 5;          // wave id 0..3 -> hidden column set
  const int lane    = tid & 31;
  const int half    = lane >> 4;
  const int rm      = lane & 15;
  const int rowbase = half * 8;
  const int jcol    = w * 16 + rm;       // hidden column this lane owns
  const bool jvalid = (jcol < HID);
  const int b0      = blockIdx.x * MTILE;

  // ---- layer-0 weight B-fragments stay register-resident (64 VGPRs) ----
  v16h B0[4][2];
  #pragma unroll
  for (int s = 0; s < 4; ++s) {
    const int row = s * HID + jcol;
    #pragma unroll
    for (int kc = 0; kc < 2; ++kc) {
      v16h f;
      #pragma unroll
      for (int e = 0; e < 16; ++e) {
        int k = kc * 32 + frag_k(half, e);
        float v = (jvalid && k < HID) ? Whh0[row * HID + k] : 0.0f;
        f[e] = (_Float16)v;
      }
      B0[s][kc] = f;
    }
  }
  // ---- layer-1 weight fragments -> LDS (gathered once) ----
  #pragma unroll
  for (int s = 0; s < 4; ++s) {
    const int row = s * HID + jcol;
    #pragma unroll
    for (int kc = 0; kc < 4; ++kc) {
      FragBits u;
      #pragma unroll
      for (int e = 0; e < 16; ++e) {
        int kg = kc * 32 + frag_k(half, e);
        float v = 0.0f;
        if (jvalid) {
          if (kg < HP) { if (kg < HID) v = Wih1[row * HID + kg]; }
          else         { int kk = kg - HP; if (kk < HID) v = Whh1[row * HID + kk]; }
        }
        u.h[e] = (_Float16)v;
      }
      const int fid  = ((w * 4 + s) << 2) + kc;
      const int base = (fid * 32 + lane) * 8;
      #pragma unroll
      for (int d = 0; d < 8; ++d) B1lds[base + d] = u.i[d];
    }
  }

  float bias0v[4], bias1v[4], wih0v[4];
  #pragma unroll
  for (int s = 0; s < 4; ++s) {
    int row = s * HID + jcol;
    bias0v[s] = jvalid ? (bih0[row] + bhh0[row]) : 0.0f;
    bias1v[s] = jvalid ? (bih1[row] + bhh1[row]) : 0.0f;
    wih0v[s]  = jvalid ? Wih0[row] : 0.0f;
  }
  const float bl = blin[0];

  for (int i = tid; i < MTILE * HSTR; i += 128) {
    hA0[i] = (_Float16)0.0f;
    hA1[i] = (_Float16)0.0f;
  }
  if (tid < HP) wlinS[tid] = (tid < HID) ? Wlin[tid] : 0.0f;
  if (tid < MTILE) xbuf[tid] = x[(b0 + tid) * TSTEPS + 0];

  float c0[8], c1[8];
  #pragma unroll
  for (int r = 0; r < 8; ++r) { c0[r] = 0.0f; c1[r] = 0.0f; }

  v16h A0a, A0b;                          // A fragments of h0_prev (zero init)
  #pragma unroll
  for (int e = 0; e < 16; ++e) { A0a[e] = (_Float16)0.0f; A0b[e] = (_Float16)0.0f; }

  __syncthreads();

  for (int t = 0; t < TSTEPS; ++t) {
    float xv[8];
    #pragma unroll
    for (int r = 0; r < 8; ++r) xv[r] = xbuf[rowbase + r];

    // ---- layer 0: gates0 = h0 @ Whh0^T + x*Wih0 + b (weights in VGPRs) ----
    v8f acc[4];
    #pragma unroll
    for (int s = 0; s < 4; ++s) {
      v8f a;
      #pragma unroll
      for (int r = 0; r < 8; ++r) a[r] = __builtin_fmaf(xv[r], wih0v[s], bias0v[s]);
      a = __builtin_amdgcn_wmma_f32_16x16x32_f16(false, A0a, false, B0[s][0], (short)0, a, false, false);
      a = __builtin_amdgcn_wmma_f32_16x16x32_f16(false, A0b, false, B0[s][1], (short)0, a, false, false);
      acc[s] = a;
    }
    #pragma unroll
    for (int r = 0; r < 8; ++r) {
      float ig = sigm(acc[0][r]);
      float fg = sigm(acc[1][r]);
      float gg = tanh_fast(acc[2][r]);
      float og = sigm(acc[3][r]);
      float c  = fg * c0[r] + ig * gg;
      c0[r] = c;
      hA0[(rowbase + r) * HSTR + jcol] = (_Float16)(og * tanh_fast(c));
    }
    __syncthreads();                                     // h0_new complete in LDS

    // ---- redistribute h (C layout -> A fragments); old h1 read here ----
    v16h A1a, A1b, A1c, A1d;
    #pragma unroll
    for (int e = 0; e < 16; ++e) {
      int k = frag_k(half, e);
      A1a[e] = hA0[rm * HSTR + k];
      A1b[e] = hA0[rm * HSTR + 32 + k];
      A1c[e] = hA1[rm * HSTR + k];
      A1d[e] = hA1[rm * HSTR + 32 + k];
    }
    __syncthreads();                                     // old h1 fully read

    // ---- layer 1: gates1 = [h0_new | h1] @ [Wih1|Whh1]^T + b ----
    #pragma unroll
    for (int s = 0; s < 4; ++s) {
      const int fid = (w * 4 + s) << 2;
      v16h bf0 = load_frag_lds(B1lds, fid + 0, lane);
      v16h bf1 = load_frag_lds(B1lds, fid + 1, lane);
      v16h bf2 = load_frag_lds(B1lds, fid + 2, lane);
      v16h bf3 = load_frag_lds(B1lds, fid + 3, lane);
      v8f a;
      #pragma unroll
      for (int r = 0; r < 8; ++r) a[r] = bias1v[s];
      a = __builtin_amdgcn_wmma_f32_16x16x32_f16(false, A1a, false, bf0, (short)0, a, false, false);
      a = __builtin_amdgcn_wmma_f32_16x16x32_f16(false, A1b, false, bf1, (short)0, a, false, false);
      a = __builtin_amdgcn_wmma_f32_16x16x32_f16(false, A1c, false, bf2, (short)0, a, false, false);
      a = __builtin_amdgcn_wmma_f32_16x16x32_f16(false, A1d, false, bf3, (short)0, a, false, false);
      acc[s] = a;
    }
    #pragma unroll
    for (int r = 0; r < 8; ++r) {
      float ig = sigm(acc[0][r]);
      float fg = sigm(acc[1][r]);
      float gg = tanh_fast(acc[2][r]);
      float og = sigm(acc[3][r]);
      float c  = fg * c1[r] + ig * gg;
      c1[r] = c;
      hA1[(rowbase + r) * HSTR + jcol] = (_Float16)(og * tanh_fast(c));
    }
    // stage next timestep's x (single barrier covers both hand-offs)
    if (tid < MTILE && (t + 1) < TSTEPS) xbuf[tid] = x[(b0 + tid) * TSTEPS + t + 1];
    __syncthreads();                                     // h1_new + xbuf ready

    // ---- output head: out[b,t] = h1 . Wlin + b_lin ----
    if (tid < MTILE) {
      float acco = bl;
      #pragma unroll 8
      for (int c = 0; c < HP; ++c)
        acco += (float)hA1[tid * HSTR + c] * wlinS[c];
      out[(b0 + tid) * TSTEPS + t] = acco;
    }

    // h0_new's A fragments become next step's layer-0 A operand
    A0a = A1a;
    A0b = A1b;
  }
}

extern "C" void kernel_launch(void* const* d_in, const int* in_sizes, int n_in,
                              void* d_out, int out_size, void* d_ws, size_t ws_size,
                              hipStream_t stream) {
  const float* x    = (const float*)d_in[0];
  const float* Wih0 = (const float*)d_in[1];
  const float* Whh0 = (const float*)d_in[2];
  const float* bih0 = (const float*)d_in[3];
  const float* bhh0 = (const float*)d_in[4];
  const float* Wih1 = (const float*)d_in[5];
  const float* Whh1 = (const float*)d_in[6];
  const float* bih1 = (const float*)d_in[7];
  const float* bhh1 = (const float*)d_in[8];
  const float* Wlin = (const float*)d_in[9];
  const float* blin = (const float*)d_in[10];
  float* out = (float*)d_out;

  dim3 grid(BATCH / MTILE);        // 64 independent batch tiles
  dim3 block(128);                 // 4 wave32 -> 4 hidden-column sets
  size_t dyn_lds = 64u * 32u * 8u * sizeof(int);   // 64KB layer-1 fragments
  hipLaunchKernelGGL(lstm2_wmma_kernel, grid, block, dyn_lds, stream,
                     x, Wih0, Whh0, bih0, bhh0, Wih1, Whh1, bih1, bhh1,
                     Wlin, blin, out);
}
